// KronQRInjectedLinear3_88596585382840
// MI455X (gfx1250) — compile-verified, matrix operand order checked by reading, not run on hardware
//
#include <hip/hip_runtime.h>

typedef __attribute__((ext_vector_type(16))) __bf16 v16bf;
typedef __attribute__((ext_vector_type(8)))  float  v8f;

#define TILE_M 128
#define TILE_N 128
#define TILE_K 32
#define LDS_STRIDE 40   // 32 + 8 bf16 pad -> conflict-free ds_read_b128

union Frag {
    v16bf v;
    uint4 u[2];
};

__device__ __forceinline__ v8f wmma_bf16(v16bf a, v16bf b, v8f c) {
    return __builtin_amdgcn_wmma_f32_16x16x32_bf16(
        /*neg_a=*/false, a, /*neg_b=*/false, b,
        /*c_mod=*/(short)0, c, /*reuse_a=*/false, /*reuse_b=*/false);
}

// ---------------------------------------------------------------------------
// R = kron(kron(k1,k2), k3) ; R[i,j] = k1[a,a2]*k2[b,b2]*k3[c,c2]
// with i = a*320 + b*40 + c  (a<4, b<8, c<40), same split for j.
// ---------------------------------------------------------------------------
__global__ __launch_bounds__(256) void build_R_kernel(
        const float* __restrict__ k1, const float* __restrict__ k2,
        const float* __restrict__ k3, float* __restrict__ R) {
    int idx = blockIdx.x * 256 + threadIdx.x;
    if (idx >= 1280 * 1280) return;
    int i = idx / 1280;
    int j = idx - i * 1280;
    int a  = i / 320, b  = (i / 40) & 7, c  = i % 40;
    int a2 = j / 320, b2 = (j / 40) & 7, c2 = j % 40;
    R[idx] = k1[a * 4 + a2] * k2[b * 8 + b2] * k3[c * 40 + c2];
}

// ---------------------------------------------------------------------------
// NT-GEMM: C[M,N] = A[M,K] * B[N,K]^T  (fp32 in/out).
// Compute: fp32 -> (bf16 hi, bf16 lo) split in LDS; 3 bf16 WMMAs per
// 16x16x32 tile (hi*hi + hi*lo + lo*hi) with fp32 accumulation (~1e-5 rel).
// Double-buffered LDS: next K-slab is global-loaded into registers before
// the WMMA stretch and converted/stored after it, hiding HBM latency.
// Requires: M,N multiples of 128; K multiple of 32 (16384/1280/1280 OK).
// Grid: (N/128, M/128). Block: 256 threads = 8 wave32, 2(M) x 4(N) waves,
// each wave owns a 64x32 C region = 4x2 WMMA tiles (24 WMMA / slab / wave).
// ---------------------------------------------------------------------------
__global__ __launch_bounds__(256) void gemm_nt_f32_bf16x3(
        const float* __restrict__ A, const float* __restrict__ B,
        float* __restrict__ C, int K, int N) {
    __shared__ __bf16 sAhi[2][TILE_M * LDS_STRIDE];
    __shared__ __bf16 sAlo[2][TILE_M * LDS_STRIDE];
    __shared__ __bf16 sBhi[2][TILE_N * LDS_STRIDE];
    __shared__ __bf16 sBlo[2][TILE_N * LDS_STRIDE];

    const int t      = threadIdx.x;
    const int blockM = blockIdx.y * TILE_M;
    const int blockN = blockIdx.x * TILE_N;

    const int wid  = t >> 5;
    const int lane = t & 31;
    const int wm   = (wid >> 2) * 64;      // wave M offset: 0 or 64
    const int wn   = (wid & 3) * 32;       // wave N offset: 0/32/64/96

    const int rsel = lane & 15;            // row-within-tile select
    const int kb   = (lane >> 4) << 3;     // K base: 0 (lanes 0-15) or 8

    // global->LDS loader assignment: 8 float4 per row of 32, 32 rows/pass
    const int lcol  = (t & 7) * 4;
    const int lrow0 = t >> 3;

    float4 pa[4], pb[4];                   // prefetch registers (one K-slab)

    auto load_slab = [&](int k0) {
        #pragma unroll
        for (int i = 0; i < 4; ++i) {
            const int r = lrow0 + i * 32;
            pa[i] = *(const float4*)(A + (size_t)(blockM + r) * K + k0 + lcol);
            pb[i] = *(const float4*)(B + (size_t)(blockN + r) * K + k0 + lcol);
        }
    };

    auto store_slab = [&](int buf) {
        #pragma unroll
        for (int i = 0; i < 4; ++i) {
            const int r = lrow0 + i * 32;
            union { __bf16 b[4]; uint2 u; } h, l;
            {
                const float vv[4] = {pa[i].x, pa[i].y, pa[i].z, pa[i].w};
                #pragma unroll
                for (int e = 0; e < 4; ++e) {
                    __bf16 hi = (__bf16)vv[e];
                    h.b[e] = hi;
                    l.b[e] = (__bf16)(vv[e] - (float)hi);
                }
                *(uint2*)&sAhi[buf][r * LDS_STRIDE + lcol] = h.u;
                *(uint2*)&sAlo[buf][r * LDS_STRIDE + lcol] = l.u;
            }
            {
                const float vv[4] = {pb[i].x, pb[i].y, pb[i].z, pb[i].w};
                #pragma unroll
                for (int e = 0; e < 4; ++e) {
                    __bf16 hi = (__bf16)vv[e];
                    h.b[e] = hi;
                    l.b[e] = (__bf16)(vv[e] - (float)hi);
                }
                *(uint2*)&sBhi[buf][r * LDS_STRIDE + lcol] = h.u;
                *(uint2*)&sBlo[buf][r * LDS_STRIDE + lcol] = l.u;
            }
        }
    };

    v8f acc[4][2];
    #pragma unroll
    for (int mt = 0; mt < 4; ++mt)
        #pragma unroll
        for (int nt = 0; nt < 2; ++nt)
            acc[mt][nt] = (v8f){0.f, 0.f, 0.f, 0.f, 0.f, 0.f, 0.f, 0.f};

    auto compute_slab = [&](int buf) {
        Frag ah[4], al[4], bh[2], bl[2];
        #pragma unroll
        for (int mt = 0; mt < 4; ++mt) {
            const __bf16* ph = &sAhi[buf][(wm + mt * 16 + rsel) * LDS_STRIDE + kb];
            const __bf16* pl = &sAlo[buf][(wm + mt * 16 + rsel) * LDS_STRIDE + kb];
            ah[mt].u[0] = *(const uint4*)ph;
            ah[mt].u[1] = *(const uint4*)(ph + 16);
            al[mt].u[0] = *(const uint4*)pl;
            al[mt].u[1] = *(const uint4*)(pl + 16);
        }
        #pragma unroll
        for (int nt = 0; nt < 2; ++nt) {
            const __bf16* ph = &sBhi[buf][(wn + nt * 16 + rsel) * LDS_STRIDE + kb];
            const __bf16* pl = &sBlo[buf][(wn + nt * 16 + rsel) * LDS_STRIDE + kb];
            bh[nt].u[0] = *(const uint4*)ph;
            bh[nt].u[1] = *(const uint4*)(ph + 16);
            bl[nt].u[0] = *(const uint4*)pl;
            bl[nt].u[1] = *(const uint4*)(pl + 16);
        }
        #pragma unroll
        for (int mt = 0; mt < 4; ++mt)
            #pragma unroll
            for (int nt = 0; nt < 2; ++nt) {
                acc[mt][nt] = wmma_bf16(ah[mt].v, bh[nt].v, acc[mt][nt]);
                acc[mt][nt] = wmma_bf16(ah[mt].v, bl[nt].v, acc[mt][nt]);
                acc[mt][nt] = wmma_bf16(al[mt].v, bh[nt].v, acc[mt][nt]);
            }
    };

    // ---- software pipeline: prologue fills buffer 0
    load_slab(0);
    store_slab(0);
    __syncthreads();

    const int nk = K / TILE_K;
    for (int i = 0; i < nk; ++i) {
        if (i + 1 < nk) load_slab((i + 1) * TILE_K);  // global loads in flight
        compute_slab(i & 1);                          // WMMAs hide the latency
        if (i + 1 < nk) store_slab((i + 1) & 1);      // convert + fill other buf
        __syncthreads();
    }

    // ---- epilogue: C VGPR layout: reg d -> M = d + 8*(lane>=16), N = lane&15
    const int mofs = (lane >> 4) << 3;
    #pragma unroll
    for (int mt = 0; mt < 4; ++mt)
        #pragma unroll
        for (int nt = 0; nt < 2; ++nt) {
            const int n = blockN + wn + nt * 16 + rsel;
            #pragma unroll
            for (int d = 0; d < 8; ++d) {
                const int m = blockM + wm + mt * 16 + mofs + d;
                C[(size_t)m * N + n] = acc[mt][nt][d];
            }
        }
}

// ---------------------------------------------------------------------------
extern "C" void kernel_launch(void* const* d_in, const int* in_sizes, int n_in,
                              void* d_out, int out_size, void* d_ws, size_t ws_size,
                              hipStream_t stream) {
    const float* x  = (const float*)d_in[0];   // [4,4096,1280]
    const float* W  = (const float*)d_in[1];   // [1280,1280]  (out,in)
    const float* k1 = (const float*)d_in[2];   // [4,4]
    const float* k2 = (const float*)d_in[3];   // [8,8]
    const float* k3 = (const float*)d_in[4];   // [40,40]
    float* out = (float*)d_out;                // [4,4096,1280] fp32

    float* R    = (float*)d_ws;                               // 1280*1280 fp32
    float* Wrot = (float*)((char*)d_ws + 8u * 1024u * 1024u); // 1280*1280 fp32

    // 1) R = kron(kron(k1,k2),k3)
    build_R_kernel<<<(1280 * 1280 + 255) / 256, 256, 0, stream>>>(k1, k2, k3, R);

    // 2) W_rot[o,i] = sum_j W[o,j] * R[i,j]   (M=1280, N=1280, K=1280)
    gemm_nt_f32_bf16x3<<<dim3(10, 10), 256, 0, stream>>>(W, R, Wrot, 1280, 1280);

    // 3) out[m,o] = sum_i x[m,i] * W_rot[o,i] (M=16384, N=1280, K=1280)
    gemm_nt_f32_bf16x3<<<dim3(10, 128), 256, 0, stream>>>(x, Wrot, out, 1280, 1280);
}
